// Feature_learning_layer_2886218023203
// MI455X (gfx1250) — compile-verified, matrix operand order checked by reading, not run on hardware
//
#include <hip/hip_runtime.h>
#include <hip/hip_bf16.h>

// ---------------------------------------------------------------------------
// Problem constants (from reference): B=8, N=256, F=64, C=16, K=2
// ---------------------------------------------------------------------------
#define BB 8
#define NN 256
#define FF 64
#define CC 16
#define KKC 2
#define KD 160              // 2F+C = 144 padded to 5*32

typedef __bf16 bf16;
typedef bf16  v16bf  __attribute__((ext_vector_type(16)));
typedef bf16  bf16x8 __attribute__((ext_vector_type(8)));
typedef float v8f    __attribute__((ext_vector_type(8)));

static __device__ __forceinline__ float frelu(float x) { return x > 0.f ? x : 0.f; }

// Load one 16-lane-value fragment (A or B operand) from a row-major bf16 LDS
// matrix. `p` already points at element [row][kbase]. Per the CDNA5 layout,
// lane half h reads K = h*8..h*8+7 (vector elems 0..7) and K = 16+h*8..+7
// (vector elems 8..15), i.e. two contiguous 16-byte chunks.
static __device__ __forceinline__ v16bf ld_frag(const bf16* p, int half) {
  const bf16x8 c0 = *(const bf16x8*)(p + half * 8);
  const bf16x8 c1 = *(const bf16x8*)(p + half * 8 + 16);
  v16bf r;
#pragma unroll
  for (int e = 0; e < 8; ++e) { r[e] = c0[e]; r[e + 8] = c1[e]; }
  return r;
}

static __device__ __forceinline__ v8f wmma_bf16(v16bf a, v16bf b, v8f c) {
  return __builtin_amdgcn_wmma_f32_16x16x32_bf16(false, a, false, b, (short)0, c,
                                                 false, false);
}

static __device__ __forceinline__ void wave_fence() {
  __builtin_amdgcn_wave_barrier();
  asm volatile("" ::: "memory");
}

// ---------------------------------------------------------------------------
// K0: per-batch mean of the input coord over (n,c); also zero mean1 accum.
// ---------------------------------------------------------------------------
__global__ __launch_bounds__(256) void mean_kernel(const float* __restrict__ coord,
                                                   float* __restrict__ mean0,
                                                   float* __restrict__ mean1acc) {
  int b = blockIdx.x, t = threadIdx.x;
  __shared__ float s[3];
  if (t < 3) s[t] = 0.f;
  __syncthreads();
  float p0 = 0.f, p1 = 0.f, p2 = 0.f;
  for (int idx = t; idx < NN * CC; idx += 256) {
    const float* p = coord + ((size_t)b * NN * CC + idx) * 3;
    p0 += p[0]; p1 += p[1]; p2 += p[2];
  }
  atomicAdd(&s[0], p0); atomicAdd(&s[1], p1); atomicAdd(&s[2], p2);
  __syncthreads();
  if (t < 3) {
    mean0[b * 3 + t] = s[t] * (1.f / (NN * CC));
    mean1acc[b * 3 + t] = 0.f;
  }
}

// ---------------------------------------------------------------------------
// K1: edge kernel — the big WMMA workhorse. One workgroup per (b, i).
//   produces  wsAgg[b,i,0:64]   = sum_{j!=i} edge_feat[b,i,j,:]
//             wsCAgg[b,i,c,d]   = sum_j factors[b,i,j,c]*(coord_i-coord_j)[c,d]
// ---------------------------------------------------------------------------
__global__ __launch_bounds__(256) void edge_kernel(
    const float* __restrict__ h, const float* __restrict__ coord,
    const float* __restrict__ category,
    const float* __restrict__ We1g, const float* __restrict__ bE1g,
    const float* __restrict__ We2g, const float* __restrict__ bE2g,
    const float* __restrict__ Wc1g, const float* __restrict__ bC1g,
    const float* __restrict__ Wc2g, const float* __restrict__ bC2g,
    const float* __restrict__ Wf1g, const float* __restrict__ bF1g,
    const float* __restrict__ Wf2g, const float* __restrict__ bF2g,
    float* __restrict__ wsAgg, float* __restrict__ wsCAgg) {
  extern __shared__ char smem[];
  // bf16 regions (all 16B aligned)
  bf16* hL    = (bf16*)smem;            // [256][64]
  bf16* we1L  = hL + NN * FF;           // [64][160]
  bf16* wc1L  = we1L + FF * KD;         // [2][64][160]
  bf16* we2L  = wc1L + 2 * FF * KD;     // [64][64]
  bf16* wc2L  = we2L + FF * FF;         // [2][16][64]
  bf16* distL = wc2L + 2 * CC * FF;     // [256][16]
  float* fr   = (float*)(distL + NN * CC);
  float* cI   = fr;                     // 48
  float* bE1  = cI + 48;                // 64
  float* bE2  = bE1 + 64;               // 64
  float* bC1  = bE2 + 64;               // 128
  float* bC2  = bC1 + 128;              // 32
  float* wF1  = bC2 + 32;               // 256
  float* wF2  = wF1 + 256;              // 256
  float* bF1  = wF2 + 256;              // 16
  float* bF2  = bF1 + 16;               // 16
  float* aggL = bF2 + 16;               // 64
  float* cAggL= aggL + 64;              // 48
  bf16* wscr  = (bf16*)(cAggL + 48);    // 8 waves * 1024 bf16 (2KB each)

  const int tid = threadIdx.x;
  const int b = blockIdx.y, i = blockIdx.x;
  const int lane = tid & 31, wave = tid >> 5;
  const int half = lane >> 4, lcol = lane & 15, lrow = lane & 15;

  if (tid == 0) __builtin_prefetch(h + (size_t)b * NN * FF, 0, 3);

  // ---- phase 0: cooperative staging ----
  for (int idx = tid; idx < NN * FF; idx += 256)
    hL[idx] = (bf16)h[(size_t)b * NN * FF + idx];
  for (int idx = tid; idx < FF * KD; idx += 256) {
    int r = idx / KD, c = idx % KD;
    we1L[idx] = (bf16)(c < 144 ? We1g[r * 144 + c] : 0.f);
  }
  for (int idx = tid; idx < 2 * FF * KD; idx += 256) {
    int k = idx / (FF * KD);
    int rem = idx - k * (FF * KD);
    int r = rem / KD, c = rem % KD;
    wc1L[idx] = (bf16)(c < 144 ? Wc1g[(k * FF + r) * 144 + c] : 0.f);
  }
  for (int idx = tid; idx < FF * FF; idx += 256) we2L[idx] = (bf16)We2g[idx];
  for (int idx = tid; idx < 2 * CC * FF; idx += 256) wc2L[idx] = (bf16)Wc2g[idx];
  if (tid < 48)  cI[tid]  = coord[((size_t)(b * NN + i) * CC) * 3 + tid];
  if (tid < 64)  { bE1[tid] = bE1g[tid]; bE2[tid] = bE2g[tid]; }
  if (tid < 128) bC1[tid] = bC1g[tid];
  if (tid < 32)  bC2[tid] = bC2g[tid];
  if (tid < 256) { wF1[tid] = Wf1g[tid]; wF2[tid] = Wf2g[tid]; }
  if (tid < 16)  { bF1[tid] = bF1g[tid]; bF2[tid] = bF2g[tid]; }
  if (tid < 64)  aggL[tid] = 0.f;
  if (tid < 48)  cAggL[tid] = 0.f;
  __syncthreads();
  { // dist[i, j=tid, c] for all c
    const float* cj = coord + ((size_t)(b * NN + tid) * CC) * 3;
#pragma unroll
    for (int c = 0; c < CC; ++c) {
      float dx = cI[c * 3 + 0] - cj[c * 3 + 0];
      float dy = cI[c * 3 + 1] - cj[c * 3 + 1];
      float dz = cI[c * 3 + 2] - cj[c * 3 + 2];
      distL[tid * CC + c] = (bf16)sqrtf(dx * dx + dy * dy + dz * dz);
    }
  }
  __syncthreads();

  // ---- phase 1: per-wave j-tiles ----
  bf16* myScr = wscr + wave * 1024;     // 2KB wave scratch
  float* fpS = (float*)myScr;           // 256 f32
  float* f2S = fpS + 256;               // 256 f32
  const float* catp = category + (size_t)(b * NN + i) * NN * KKC;

  for (int jt = wave; jt < 16; jt += 8) {
    const int jr = jt * 16 + lrow;
    // A fragments for the K=160 edge_h GEMMs (shared by E1 / cat0 / cat1)
    v16bf a[5];
    a[0] = ld_frag(hL + i * FF + 0, half);    // h_i cols 0..31 (broadcast rows)
    a[1] = ld_frag(hL + i * FF + 32, half);   // h_i cols 32..63
    a[2] = ld_frag(hL + jr * FF + 0, half);   // h_j cols 64..95
    a[3] = ld_frag(hL + jr * FF + 32, half);  // h_j cols 96..127
    {
      bf16x8 c0 = *(const bf16x8*)(distL + jr * CC + half * 8);
      v16bf t;
#pragma unroll
      for (int e = 0; e < 8; ++e) { t[e] = c0[e]; t[e + 8] = (bf16)0.f; }
      a[4] = t;                               // dist cols 128..143 (+pad)
    }

    v8f e1[4], cA[2][4];
#pragma unroll
    for (int nt = 0; nt < 4; ++nt) {
      e1[nt] = (v8f)0.f; cA[0][nt] = (v8f)0.f; cA[1][nt] = (v8f)0.f;
    }
#pragma unroll
    for (int ks = 0; ks < 5; ++ks) {
#pragma unroll
      for (int nt = 0; nt < 4; ++nt) {
        v16bf bf;
        bf = ld_frag(we1L + (nt * 16 + lcol) * KD + ks * 32, half);
        e1[nt] = wmma_bf16(a[ks], bf, e1[nt]);
        bf = ld_frag(wc1L + (nt * 16 + lcol) * KD + ks * 32, half);
        cA[0][nt] = wmma_bf16(a[ks], bf, cA[0][nt]);
        bf = ld_frag(wc1L + FF * KD + (nt * 16 + lcol) * KD + ks * 32, half);
        cA[1][nt] = wmma_bf16(a[ks], bf, cA[1][nt]);
      }
    }

    // E1 -> relu -> wave scratch as A for E2
#pragma unroll
    for (int nt = 0; nt < 4; ++nt) {
      float bb = bE1[nt * 16 + lcol];
#pragma unroll
      for (int r = 0; r < 8; ++r)
        myScr[(r + 8 * half) * FF + nt * 16 + lcol] = (bf16)frelu(e1[nt][r] + bb);
    }
    wave_fence();

    v8f e2[4];
#pragma unroll
    for (int nt = 0; nt < 4; ++nt) e2[nt] = (v8f)0.f;
#pragma unroll
    for (int ks = 0; ks < 2; ++ks) {
      v16bf av = ld_frag(myScr + lrow * FF + ks * 32, half);
#pragma unroll
      for (int nt = 0; nt < 4; ++nt) {
        v16bf bf = ld_frag(we2L + (nt * 16 + lcol) * FF + ks * 32, half);
        e2[nt] = wmma_bf16(av, bf, e2[nt]);
      }
    }
    // agg accumulation (exclude self edge j==i)
#pragma unroll
    for (int nt = 0; nt < 4; ++nt) {
      float bb = bE2[nt * 16 + lcol];
      float s = 0.f;
#pragma unroll
      for (int r = 0; r < 8; ++r) {
        int j = jt * 16 + r + 8 * half;
        float v = frelu(e2[nt][r] + bb);
        s += (j != i) ? v : 0.f;
      }
      atomicAdd(&aggL[nt * 16 + lcol], s);
    }

    // category paths: relu -> scratch -> x W_cat2 -> weighted sum
    v8f fp = (v8f)0.f;
#pragma unroll
    for (int kc = 0; kc < KKC; ++kc) {
      wave_fence();
#pragma unroll
      for (int nt = 0; nt < 4; ++nt) {
        float bb = bC1[kc * 64 + nt * 16 + lcol];
#pragma unroll
        for (int r = 0; r < 8; ++r)
          myScr[(r + 8 * half) * FF + nt * 16 + lcol] =
              (bf16)frelu(cA[kc][nt][r] + bb);
      }
      wave_fence();
      v8f cd = (v8f)0.f;
#pragma unroll
      for (int ks = 0; ks < 2; ++ks) {
        v16bf av = ld_frag(myScr + lrow * FF + ks * 32, half);
        v16bf bf = ld_frag(wc2L + kc * CC * FF + lcol * FF + ks * 32, half);
        cd = wmma_bf16(av, bf, cd);
      }
      float bb2 = bC2[kc * 16 + lcol];
#pragma unroll
      for (int r = 0; r < 8; ++r) {
        int j = jt * 16 + r + 8 * half;
        float catv = catp[j * KKC + kc];
        fp[r] += catv * frelu(cd[r] + bb2);
      }
    }

    // spill fp tile [16j][16c] and run the tiny C=16 relu MLP per row
    wave_fence();
#pragma unroll
    for (int r = 0; r < 8; ++r) fpS[(r + 8 * half) * 16 + lcol] = fp[r];
    wave_fence();
    if (lane < 16) {
      float fv[16], t1[16];
#pragma unroll
      for (int c = 0; c < 16; ++c) fv[c] = fpS[lane * 16 + c];
#pragma unroll
      for (int o = 0; o < 16; ++o) {
        float s = bF1[o];
#pragma unroll
        for (int c = 0; c < 16; ++c) s += wF1[o * 16 + c] * fv[c];
        t1[o] = frelu(s);
      }
#pragma unroll
      for (int c = 0; c < 16; ++c) {
        float s = bF2[c];
#pragma unroll
        for (int o = 0; o < 16; ++o) s += wF2[c * 16 + o] * t1[o];
        f2S[lane * 16 + c] = frelu(s);
      }
    }
    wave_fence();
    // coordAgg[c,d] += sum_j factors[j,c] * (coord_i - coord_j)[c,d]
    {
      int c = lcol;
#pragma unroll
      for (int d = 0; d < 3; ++d) {
        float s = 0.f;
#pragma unroll
        for (int jl = 0; jl < 8; ++jl) {
          int j = jt * 16 + half * 8 + jl;
          float f = f2S[(half * 8 + jl) * 16 + c];
          float cj = coord[(((size_t)(b * NN + j)) * CC + c) * 3 + d];
          s += f * (cI[c * 3 + d] - cj);
        }
        atomicAdd(&cAggL[c * 3 + d], s);
      }
    }
    wave_fence();
  }
  __syncthreads();

  // ---- phase 2: write results ----
  if (tid < 64) wsAgg[(size_t)(b * NN + i) * 64 + tid] = aggL[tid];
  if (tid < 48) wsCAgg[(size_t)(b * NN + i) * 48 + tid] = cAggL[tid];
}

// ---------------------------------------------------------------------------
// K2: attention + coordinate mid-update; accumulate new per-batch mean.
// ---------------------------------------------------------------------------
__global__ __launch_bounds__(256) void coord_mid_kernel(
    const float* __restrict__ h, const float* __restrict__ coord,
    const float* __restrict__ vel, const float* __restrict__ Wcv,
    const float* __restrict__ Wqm, const float* __restrict__ bqm,
    const float* __restrict__ mean0, const float* __restrict__ wsCAgg,
    float* __restrict__ coord2, float* __restrict__ mean1acc) {
  int b = blockIdx.x, n = threadIdx.x;
  size_t bn = (size_t)b * NN + n;
  const float* hr = h + bn * FF;
  float att[CC];
#pragma unroll
  for (int c = 0; c < CC; ++c) {
    float s = bqm[c];
#pragma unroll
    for (int f = 0; f < FF; ++f) s += Wqm[c * FF + f] * hr[f];
    att[c] = frelu(s);
  }
  float vl[CC][3];
#pragma unroll
  for (int c = 0; c < CC; ++c)
#pragma unroll
    for (int d = 0; d < 3; ++d) vl[c][d] = vel[(bn * CC + c) * 3 + d];
  float m0[3] = {mean0[b * 3], mean0[b * 3 + 1], mean0[b * 3 + 2]};
  float sums[3] = {0.f, 0.f, 0.f};
#pragma unroll
  for (int c = 0; c < CC; ++c) {
#pragma unroll
    for (int d = 0; d < 3; ++d) {
      size_t idx = (bn * CC + c) * 3 + d;
      float cv = coord[idx];
      float velsum = 0.f;
#pragma unroll
      for (int c2 = 0; c2 < CC; ++c2) velsum += Wcv[c * CC + c2] * vl[c2][d];
      float val = att[c] * (cv - m0[d]) + cv + wsCAgg[bn * 48 + c * 3 + d] + velsum;
      coord2[idx] = val;
      sums[d] += val;
    }
  }
  atomicAdd(&mean1acc[b * 3 + 0], sums[0]);
  atomicAdd(&mean1acc[b * 3 + 1], sums[1]);
  atomicAdd(&mean1acc[b * 3 + 2], sums[2]);
}

// ---------------------------------------------------------------------------
// K3: non-linear equivariant coordinate finalize.
// ---------------------------------------------------------------------------
__global__ __launch_bounds__(256) void coord_fin_kernel(
    const float* __restrict__ coord2, const float* __restrict__ mean1acc,
    const float* __restrict__ Wql, const float* __restrict__ Wkl,
    float* __restrict__ coordOut) {
  int b = blockIdx.x, n = threadIdx.x;
  size_t bn = (size_t)b * NN + n;
  float cm[3];
#pragma unroll
  for (int d = 0; d < 3; ++d) cm[d] = mean1acc[b * 3 + d] * (1.f / (NN * CC));
  float cc[CC][3];
#pragma unroll
  for (int c = 0; c < CC; ++c)
#pragma unroll
    for (int d = 0; d < 3; ++d) cc[c][d] = coord2[(bn * CC + c) * 3 + d] - cm[d];
#pragma unroll
  for (int o = 0; o < CC; ++o) {
    float q[3] = {0.f, 0.f, 0.f}, k[3] = {0.f, 0.f, 0.f};
#pragma unroll
    for (int c = 0; c < CC; ++c) {
      float wq = Wql[o * CC + c], wk = Wkl[o * CC + c];
#pragma unroll
      for (int d = 0; d < 3; ++d) { q[d] += wq * cc[c][d]; k[d] += wk * cc[c][d]; }
    }
    float prod = q[0] * k[0] + q[1] * k[1] + q[2] * k[2];
    float kns = k[0] * k[0] + k[1] * k[1] + k[2] * k[2];
#pragma unroll
    for (int d = 0; d < 3; ++d) {
      float out = (prod >= 0.f) ? q[d] : (q[d] - prod / (kns + 1e-4f) * k[d]);
      coordOut[(bn * CC + o) * 3 + d] = out + cm[d];
    }
  }
}

// ---------------------------------------------------------------------------
// K4: node MLP with WMMA. M = B*N = 2048 rows, A = [h | agg], K = 128.
// 16 blocks x 8 waves; each wave one 16-row m-tile, 4 n-tiles.
// ---------------------------------------------------------------------------
__global__ __launch_bounds__(256) void node_kernel(
    const float* __restrict__ h, const float* __restrict__ wsAgg,
    const float* __restrict__ Wn1g, const float* __restrict__ bN1g,
    const float* __restrict__ Wn2g, const float* __restrict__ bN2g,
    float* __restrict__ hOut) {
  extern __shared__ char smem[];
  bf16* AL   = (bf16*)smem;            // [128][128]
  bf16* wn1L = AL + 128 * 128;         // [64][128]
  bf16* wn2L = wn1L + FF * 128;        // [64][64]
  float* bN1 = (float*)(wn2L + FF * FF); // 64
  float* bN2 = bN1 + 64;               // 64
  bf16* wscr = (bf16*)(bN2 + 64);      // 8 * 1024 bf16

  const int tid = threadIdx.x;
  const int lane = tid & 31, wave = tid >> 5;
  const int half = lane >> 4, lcol = lane & 15, lrow = lane & 15;
  const int g0 = blockIdx.x * 128;

  for (int idx = tid; idx < 128 * 128; idx += 256) {
    int r = idx / 128, c = idx % 128;
    float v = (c < FF) ? h[(size_t)(g0 + r) * FF + c]
                       : wsAgg[(size_t)(g0 + r) * FF + (c - FF)];
    AL[idx] = (bf16)v;
  }
  for (int idx = tid; idx < FF * 128; idx += 256) wn1L[idx] = (bf16)Wn1g[idx];
  for (int idx = tid; idx < FF * FF; idx += 256)  wn2L[idx] = (bf16)Wn2g[idx];
  if (tid < 64) { bN1[tid] = bN1g[tid]; bN2[tid] = bN2g[tid]; }
  __syncthreads();

  bf16* myScr = wscr + wave * 1024;
  const int m0 = wave * 16;

  v8f a1[4];
#pragma unroll
  for (int nt = 0; nt < 4; ++nt) a1[nt] = (v8f)0.f;
#pragma unroll
  for (int ks = 0; ks < 4; ++ks) {
    v16bf av = ld_frag(AL + (m0 + lrow) * 128 + ks * 32, half);
#pragma unroll
    for (int nt = 0; nt < 4; ++nt) {
      v16bf bf = ld_frag(wn1L + (nt * 16 + lcol) * 128 + ks * 32, half);
      a1[nt] = wmma_bf16(av, bf, a1[nt]);
    }
  }
#pragma unroll
  for (int nt = 0; nt < 4; ++nt) {
    float bb = bN1[nt * 16 + lcol];
#pragma unroll
    for (int r = 0; r < 8; ++r)
      myScr[(r + 8 * half) * FF + nt * 16 + lcol] = (bf16)frelu(a1[nt][r] + bb);
  }
  wave_fence();
  v8f a2[4];
#pragma unroll
  for (int nt = 0; nt < 4; ++nt) a2[nt] = (v8f)0.f;
#pragma unroll
  for (int ks = 0; ks < 2; ++ks) {
    v16bf av = ld_frag(myScr + lrow * FF + ks * 32, half);
#pragma unroll
    for (int nt = 0; nt < 4; ++nt) {
      v16bf bf = ld_frag(wn2L + (nt * 16 + lcol) * FF + ks * 32, half);
      a2[nt] = wmma_bf16(av, bf, a2[nt]);
    }
  }
#pragma unroll
  for (int nt = 0; nt < 4; ++nt) {
    int ncol = nt * 16 + lcol;
    float bb = bN2[ncol];
#pragma unroll
    for (int r = 0; r < 8; ++r) {
      size_t g = (size_t)(g0 + m0 + r + 8 * half);
      hOut[g * FF + ncol] = h[g * FF + ncol] + a2[nt][r] + bb;
    }
  }
}

// ---------------------------------------------------------------------------
// Launch
// ---------------------------------------------------------------------------
extern "C" void kernel_launch(void* const* d_in, const int* in_sizes, int n_in,
                              void* d_out, int out_size, void* d_ws, size_t ws_size,
                              hipStream_t stream) {
  const float* h     = (const float*)d_in[0];
  const float* coord = (const float*)d_in[1];
  const float* vel   = (const float*)d_in[2];
  const float* cat   = (const float*)d_in[3];
  const float* Wcv   = (const float*)d_in[4];
  const float* We1   = (const float*)d_in[5];
  const float* bE1   = (const float*)d_in[6];
  const float* We2   = (const float*)d_in[7];
  const float* bE2   = (const float*)d_in[8];
  const float* Wc1   = (const float*)d_in[9];
  const float* bC1   = (const float*)d_in[10];
  const float* Wc2   = (const float*)d_in[11];
  const float* bC2   = (const float*)d_in[12];
  const float* Wf1   = (const float*)d_in[13];
  const float* bF1   = (const float*)d_in[14];
  const float* Wf2   = (const float*)d_in[15];
  const float* bF2   = (const float*)d_in[16];
  const float* Wn1   = (const float*)d_in[17];
  const float* bN1   = (const float*)d_in[18];
  const float* Wn2   = (const float*)d_in[19];
  const float* bN2   = (const float*)d_in[20];
  const float* Wql   = (const float*)d_in[21];
  const float* Wkl   = (const float*)d_in[22];
  const float* Wqm   = (const float*)d_in[23];
  const float* bqm   = (const float*)d_in[24];

  float* ws = (float*)d_ws;
  float* mean0  = ws;             // 24
  float* mean1  = ws + 24;        // 24
  float* wsAgg  = ws + 64;        // 8*256*64  = 131072
  float* wsCAgg = wsAgg + 131072; // 8*256*48  = 98304
  float* coord2 = wsCAgg + 98304; // 8*256*48  = 98304

  float* out      = (float*)d_out;
  float* hOut     = out;                 // 131072
  float* coordOut = out + 131072;        // 98304
  float* catOut   = out + 229376;        // 1048576

  // LDS sizes (bytes)
  const size_t edgeLds = (size_t)(NN * FF + FF * KD + 2 * FF * KD + FF * FF +
                                  2 * CC * FF + NN * CC) * 2 +
                         992 * 4 + 8 * 2048;     // = 135040
  const size_t nodeLds = (size_t)(128 * 128 + FF * 128 + FF * FF) * 2 +
                         128 * 4 + 8 * 2048;     // = 74240

  mean_kernel<<<BB, 256, 0, stream>>>(coord, mean0, mean1);
  edge_kernel<<<dim3(NN, BB), 256, edgeLds, stream>>>(
      h, coord, cat, We1, bE1, We2, bE2, Wc1, bC1, Wc2, bC2, Wf1, bF1, Wf2, bF2,
      wsAgg, wsCAgg);
  coord_mid_kernel<<<BB, 256, 0, stream>>>(h, coord, vel, Wcv, Wqm, bqm, mean0,
                                           wsCAgg, coord2, mean1);
  coord_fin_kernel<<<BB, 256, 0, stream>>>(coord2, mean1, Wql, Wkl, coordOut);
  node_kernel<<<2048 / 128, 256, nodeLds, stream>>>(h, wsAgg, Wn1, bN1, Wn2, bN2,
                                                    hOut);
  hipMemcpyAsync(catOut, cat, (size_t)BB * NN * NN * KKC * sizeof(float),
                 hipMemcpyDeviceToDevice, stream);
}
